// LUTGeMM_80616536146034
// MI455X (gfx1250) — 8-bit matrix operand order repaired and checked by reading, compiled, not run on hardware
//
#include <hip/hip_runtime.h>
#include <stdint.h>

typedef __attribute__((ext_vector_type(8))) int v8i;

__global__ __launch_bounds__(128)
void quant_rows_kernel(const float* __restrict__ src,
                       int8_t* __restrict__ dst,
                       int* __restrict__ sums,
                       const float* __restrict__ scale_p,
                       const float* __restrict__ zp_p) {
    const float scale = scale_p[0];
    const float zp    = zp_p[0];
    const int   row   = blockIdx.x;
    const int   t     = threadIdx.x;
    const int   base  = row * 512 + t * 4;

    float4 x = *reinterpret_cast<const float4*>(src + base);
    int q0 = (int)fminf(fmaxf(rintf(x.x / scale + zp), -128.0f), 127.0f);
    int q1 = (int)fminf(fmaxf(rintf(x.y / scale + zp), -128.0f), 127.0f);
    int q2 = (int)fminf(fmaxf(rintf(x.z / scale + zp), -128.0f), 127.0f);
    int q3 = (int)fminf(fmaxf(rintf(x.w / scale + zp), -128.0f), 127.0f);

    uint32_t packed = (uint32_t)(q0 & 0xFF)        | ((uint32_t)(q1 & 0xFF) << 8) |
                      ((uint32_t)(q2 & 0xFF) << 16) | ((uint32_t)(q3 & 0xFF) << 24);
    *reinterpret_cast<uint32_t*>(dst + base) = packed;

    int s = q0 + q1 + q2 + q3;
    #pragma unroll
    for (int off = 16; off > 0; off >>= 1)
        s += __shfl_down(s, off, 32);

    __shared__ int wpart[4];
    if ((t & 31) == 0) wpart[t >> 5] = s;
    __syncthreads();
    if (t == 0) sums[row] = wpart[0] + wpart[1] + wpart[2] + wpart[3];
}

__global__ __launch_bounds__(256)
void lutgemm_wmma_iu8_kernel(const int8_t* __restrict__ xq,
                             const int8_t* __restrict__ wq,
                             const int* __restrict__ xsum,
                             const int* __restrict__ wsum,
                             float* __restrict__ out,
                             const float* __restrict__ xs_p,
                             const float* __restrict__ xzp_p,
                             const float* __restrict__ ws_p,
                             const float* __restrict__ wzp_p) {
    __shared__ __align__(16) int8_t As[64 * 64];
    __shared__ __align__(16) int8_t Bs[64 * 64];

    const int tid     = threadIdx.x;
    const int lane    = tid & 31;
    const int wave    = tid >> 5;
    const int waveRow = wave >> 1;
    const int waveCol = wave & 1;
    const int half    = lane >> 4;
    const int l16     = lane & 15;

    const int rowBase = blockIdx.x * 64;
    const int colBase = blockIdx.y * 64;

    const int ldRow   = tid >> 2;
    const int ldChunk = (tid & 3) * 16;

    const uint32_t lds_a = (uint32_t)(uintptr_t)&As[ldRow * 64 + ldChunk];
    const uint32_t lds_b = (uint32_t)(uintptr_t)&Bs[ldRow * 64 + ldChunk];

    v8i acc0 = {};
    v8i acc1 = {};

    for (int k0 = 0; k0 < 512; k0 += 64) {
        __syncthreads();

        const uint32_t goff_a = (uint32_t)((rowBase + ldRow) * 512 + k0 + ldChunk);
        const uint32_t goff_b = (uint32_t)((colBase + ldRow) * 512 + k0 + ldChunk);
        asm volatile("global_load_async_to_lds_b128 %0, %1, %2"
                     :
                     : "v"(lds_a), "v"(goff_a), "s"(xq)
                     : "memory");
        asm volatile("global_load_async_to_lds_b128 %0, %1, %2"
                     :
                     : "v"(lds_b), "v"(goff_b), "s"(wq)
                     : "memory");
        asm volatile("s_wait_asynccnt 0x0" ::: "memory");

        __syncthreads();

        v8i a;
        {
            const int8_t* ap = &As[(waveRow * 16 + l16) * 64 + half * 16];
            #pragma unroll
            for (int g = 0; g < 4; ++g) {
                int2 d = *reinterpret_cast<const int2*>(ap + (g & 1) * 8 + (g >> 1) * 32);
                a[2 * g]     = d.x;
                a[2 * g + 1] = d.y;
            }
        }

        v8i b0, b1;
        {
            const int8_t* bp0 = &Bs[(waveCol * 32 + l16) * 64 + half * 16];
            int4 lo = *reinterpret_cast<const int4*>(bp0);
            int4 hi = *reinterpret_cast<const int4*>(bp0 + 32);
            b0[0] = lo.x; b0[1] = lo.y; b0[2] = lo.z; b0[3] = lo.w;
            b0[4] = hi.x; b0[5] = hi.y; b0[6] = hi.z; b0[7] = hi.w;

            const int8_t* bp1 = &Bs[(waveCol * 32 + 16 + l16) * 64 + half * 16];
            lo = *reinterpret_cast<const int4*>(bp1);
            hi = *reinterpret_cast<const int4*>(bp1 + 32);
            b1[0] = lo.x; b1[1] = lo.y; b1[2] = lo.z; b1[3] = lo.w;
            b1[4] = hi.x; b1[5] = hi.y; b1[6] = hi.z; b1[7] = hi.w;
        }

        acc0 = __builtin_amdgcn_wmma_i32_16x16x64_iu8(true, a, true, b0, acc0, false, false);
        acc1 = __builtin_amdgcn_wmma_i32_16x16x64_iu8(true, a, true, b1, acc1, false, false);
    }

    const float xs  = xs_p[0];
    const float xzp = xzp_p[0];
    const float ws  = ws_p[0];
    const float wzp = wzp_p[0];
    const float cterm = 512.0f * xzp * wzp;
    const float ss    = xs * ws;

    #pragma unroll
    for (int i = 0; i < 8; ++i) {
        const int gRow = rowBase + waveRow * 16 + half * 8 + i;
        const float xcorr = wzp * (float)xsum[gRow];

        const int gc0 = colBase + waveCol * 32 + l16;
        out[gRow * 256 + gc0] =
            ((float)acc0[i] - xcorr - xzp * (float)wsum[gc0] + cterm) * ss;

        const int gc1 = gc0 + 16;
        out[gRow * 256 + gc1] =
            ((float)acc1[i] - xcorr - xzp * (float)wsum[gc1] + cterm) * ss;
    }
}

extern "C" void kernel_launch(void* const* d_in, const int* in_sizes, int n_in,
                              void* d_out, int out_size, void* d_ws, size_t ws_size,
                              hipStream_t stream) {
    const float* x       = (const float*)d_in[0];
    const float* w       = (const float*)d_in[1];
    const float* x_scale = (const float*)d_in[3];
    const float* x_zp    = (const float*)d_in[4];
    const float* w_scale = (const float*)d_in[5];
    const float* w_zp    = (const float*)d_in[6];

    int8_t* xq   = (int8_t*)d_ws;
    int8_t* wq   = xq + 2048 * 512;
    int*    xsum = (int*)(wq + 256 * 512);
    int*    wsum = xsum + 2048;

    quant_rows_kernel<<<2048, 128, 0, stream>>>(x, xq, xsum, x_scale, x_zp);
    quant_rows_kernel<<<256,  128, 0, stream>>>(w, wq, wsum, w_scale, w_zp);

    lutgemm_wmma_iu8_kernel<<<dim3(32, 4), 256, 0, stream>>>(
        xq, wq, xsum, wsum, (float*)d_out, x_scale, x_zp, w_scale, w_zp);
}
